// CriticSNN_76682346103361
// MI455X (gfx1250) — matrix operand order repaired and checked by reading, not compiled, not run on hardware
//
#include <hip/hip_runtime.h>
#include <hip/hip_bf16.h>

// ---------------------------------------------------------------------------
// CDNA5 (gfx1250) fused SNN critic.
//  * Hidden-layer GEMMs (98% of FLOPs): V_WMMA_F32_16X16X32_BF16 with an
//    EXACT 3-term bf16 truncation split of the fp32 weights (spikes are {0,1},
//    exact in bf16; every partial product is exact; f32 accumulate).
//  * Input GEMM (h_in): exact fp32 V_WMMA_F32_16X16X4_F32 path.
//  * Tile staging: GLOBAL_LOAD_ASYNC_TO_LDS_B128 (ASYNCcnt) when available.
// ---------------------------------------------------------------------------

typedef float  v2f     __attribute__((ext_vector_type(2)));
typedef float  v8f     __attribute__((ext_vector_type(8)));
typedef int    v4i     __attribute__((vector_size(16)));     // b128 payload
typedef __bf16 bf16x8  __attribute__((ext_vector_type(8)));
typedef __bf16 bf16x16 __attribute__((ext_vector_type(16)));

#define BM 64          // batch rows per workgroup tile
#define BN 128         // output neurons per workgroup tile
#define BK 64          // fp32 path: K chunk staged in LDS
#define LDS_STRIDE 68  // fp32 path: 64 + 4 pad (conflict-free ds_load_b64)

#define BKH 32         // bf16 path: K chunk (one 16x16x32 k-step)
#define HSTR 40        // bf16 path LDS row stride (32 + 8 pad, 16B aligned)
#define BPLANE (BN * HSTR)

#if defined(__has_builtin)
#  if __has_builtin(__builtin_amdgcn_global_load_async_to_lds_b128)
#    define HAVE_ASYNC_B128 1
#  endif
#  if __has_builtin(__builtin_amdgcn_s_wait_asynccnt)
#    define HAVE_WAIT_ASYNC 1
#  endif
#endif
#ifndef HAVE_ASYNC_B128
#  define HAVE_ASYNC_B128 0
#endif
#ifndef HAVE_WAIT_ASYNC
#  define HAVE_WAIT_ASYNC 0
#endif

typedef __attribute__((address_space(1))) v4i as1_v4i;   // global int4
typedef __attribute__((address_space(3))) v4i as3_v4i;   // LDS int4

__device__ __forceinline__ void async_wait_zero() {
#if HAVE_ASYNC_B128
#  if HAVE_WAIT_ASYNC
    __builtin_amdgcn_s_wait_asynccnt(0);
#  else
    asm volatile("s_wait_asynccnt 0x0" ::: "memory");
#  endif
#endif
}

// 16-byte global->LDS copy: async (ASYNCcnt) when available, else via VGPRs.
__device__ __forceinline__ void copy16(const void* src, void* dst) {
#if HAVE_ASYNC_B128
    __builtin_amdgcn_global_load_async_to_lds_b128(
        (as1_v4i*)(size_t)src, (as3_v4i*)(unsigned)(size_t)dst, 0, 0);
#else
    *(v4i*)dst = *(const v4i*)src;
#endif
}

// ======================== fp32 WMMA path (h_in GEMM) ========================

template <int ROWS>
__device__ __forceinline__ void load_tile_f32(const float* __restrict__ g, int ld,
                                              int row0, int col0, int kvalid, int Ktotal,
                                              float* lds) {
    const int tid = threadIdx.x;
    const int c4  = (tid & 15) << 2;
    const int rb  = tid >> 4;
#pragma unroll
    for (int i = 0; i < ROWS / 16; ++i) {
        const int row = rb + (i << 4);
        const float* src = g + (size_t)(row0 + row) * (size_t)ld + col0;
        float* dst = lds + row * LDS_STRIDE + c4;
        if (c4 + 3 < kvalid) {
            copy16(src + c4, dst);
        } else {
            float4 v;
            v.x = (c4 + 0 < kvalid) ? src[c4 + 0] : 0.0f;
            v.y = (c4 + 1 < kvalid) ? src[c4 + 1] : 0.0f;
            v.z = (c4 + 2 < kvalid) ? src[c4 + 2] : 0.0f;
            v.w = (c4 + 3 < kvalid) ? src[c4 + 3] : 0.0f;
            *(float4*)dst = v;
        }
        if (col0 + BK < Ktotal) __builtin_prefetch(src + BK + c4, 0, 0);
    }
}

// fp32 WMMA lane layout (ISA 7.12.2): A 16x4: lane l -> M=l%16, K=2*(l/16)+v.
// B 4x16 mirrored.  C/D: VGPR r -> M=r+8*(l/16), N=l%16.
__device__ __forceinline__ void gemm_tiles_f32(const float* __restrict__ A, int lda,
                                               const float* __restrict__ W, int ldw,
                                               int K, int m0, int n0,
                                               float* ldsA, float* ldsB,
                                               v8f (&acc)[2][2]) {
    const int lane  = threadIdx.x & 31;
    const int wave  = threadIdx.x >> 5;
    const int wm    = wave & 1;
    const int wn    = wave >> 1;
    const int a0row = wm * 32 + (lane & 15);
    const int a1row = a0row + 16;
    const int b0row = wn * 32 + (lane & 15);
    const int b1row = b0row + 16;
    const int khalf = (lane >> 4) << 1;

    for (int k0 = 0; k0 < K; k0 += BK) {
        int kvalid = K - k0; if (kvalid > BK) kvalid = BK;
        __syncthreads();
        load_tile_f32<BM>(A, lda, m0, k0, kvalid, K, ldsA);
        load_tile_f32<BN>(W, ldw, n0, k0, kvalid, K, ldsB);
        async_wait_zero();
        __syncthreads();
        const int kmax = (kvalid + 3) & ~3;
        for (int kk = 0; kk < kmax; kk += 4) {
            v2f a0 = *(const v2f*)(ldsA + a0row * LDS_STRIDE + kk + khalf);
            v2f a1 = *(const v2f*)(ldsA + a1row * LDS_STRIDE + kk + khalf);
            v2f b0 = *(const v2f*)(ldsB + b0row * LDS_STRIDE + kk + khalf);
            v2f b1 = *(const v2f*)(ldsB + b1row * LDS_STRIDE + kk + khalf);
            acc[0][0] = __builtin_amdgcn_wmma_f32_16x16x4_f32(false, a0, false, b0,
                                                              (short)0, acc[0][0], false, false);
            acc[0][1] = __builtin_amdgcn_wmma_f32_16x16x4_f32(false, a0, false, b1,
                                                              (short)0, acc[0][1], false, false);
            acc[1][0] = __builtin_amdgcn_wmma_f32_16x16x4_f32(false, a1, false, b0,
                                                              (short)0, acc[1][0], false, false);
            acc[1][1] = __builtin_amdgcn_wmma_f32_16x16x4_f32(false, a1, false, b1,
                                                              (short)0, acc[1][1], false, false);
        }
    }
}

// C = A @ W^T (+bias) or C += A @ W^T   (builds h_in from state+action)
__global__ __launch_bounds__(256) void wmma_gemm_bias_kernel(
        const float* __restrict__ A, int lda,
        const float* __restrict__ W, int ldw, int K,
        const float* __restrict__ bias,
        float* __restrict__ C, int ldc, int accumulate) {
    __shared__ float ldsA[BM * LDS_STRIDE];
    __shared__ float ldsB[BN * LDS_STRIDE];
    const int m0 = blockIdx.x * BM;
    const int n0 = blockIdx.y * BN;
    v8f acc[2][2];
#pragma unroll
    for (int i = 0; i < 2; ++i)
#pragma unroll
        for (int j = 0; j < 2; ++j) acc[i][j] = (v8f){0.f,0.f,0.f,0.f,0.f,0.f,0.f,0.f};
    gemm_tiles_f32(A, lda, W, ldw, K, m0, n0, ldsA, ldsB, acc);

    const int lane = threadIdx.x & 31;
    const int wave = threadIdx.x >> 5;
    const int wm   = wave & 1;
    const int wn   = wave >> 1;
#pragma unroll
    for (int ms = 0; ms < 2; ++ms) {
        const int mb = m0 + wm * 32 + ms * 16 + ((lane >> 4) << 3);
#pragma unroll
        for (int ns = 0; ns < 2; ++ns) {
            const int nc = n0 + wn * 32 + ns * 16 + (lane & 15);
            const float bi = bias ? bias[nc] : 0.0f;
#pragma unroll
            for (int r = 0; r < 8; ++r) {
                const size_t idx = (size_t)(mb + r) * (size_t)ldc + nc;
                float v = acc[ms][ns][r] + bi;
                if (accumulate) v += C[idx];
                C[idx] = v;
            }
        }
    }
}

// ================== exact 3-term bf16 split of fp32 weights =================
// W = W0 + W1 + W2, Wq = bf16 truncation of the running residual. fp32 has a
// 24-bit significand; each truncation peels >=8 leading bits, so the third
// residual fits bf16 exactly -> lossless decomposition for these magnitudes.
__global__ __launch_bounds__(256) void split_w_kernel(
        const float* __restrict__ W, unsigned short* __restrict__ P,
        size_t HH, size_t total) {
    const size_t stride = (size_t)gridDim.x * blockDim.x;
    for (size_t j = (size_t)blockIdx.x * blockDim.x + threadIdx.x; j < total; j += stride) {
        const size_t L = j / HH, idx = j - L * HH;
        const float w = W[j];
        const unsigned t0 = __float_as_uint(w) & 0xFFFF0000u;
        const float r1 = w - __uint_as_float(t0);
        const unsigned t1 = __float_as_uint(r1) & 0xFFFF0000u;
        const float r2 = r1 - __uint_as_float(t1);
        const unsigned t2 = __float_as_uint(r2) & 0xFFFF0000u;
        unsigned short* base = P + L * 3 * HH + idx;
        base[0]      = (unsigned short)(t0 >> 16);
        base[HH]     = (unsigned short)(t1 >> 16);
        base[2 * HH] = (unsigned short)(t2 >> 16);
    }
}

// ==================== bf16 WMMA path (hidden layers) ========================
// 16-bit A 16x32 layout (ISA 7.12.2): lane l -> M=l%16, half=l/16;
//   VGPR0..3 = K = 8*half+0..7 (pairs), VGPR4..7 = K = 16+8*half+0..7.
// B 32x16: lane l -> N=l%16; same K order as A (8*half+0..7, 16+8*half+0..7).
__device__ __forceinline__ bf16x16 load_a_frag(const unsigned short* lds, int row, int half) {
    const unsigned short* p = lds + row * HSTR;
    bf16x8 lo = *(const bf16x8*)(p + 8 * half);
    bf16x8 hi = *(const bf16x8*)(p + 16 + 8 * half);
    return __builtin_shufflevector(lo, hi, 0,1,2,3,4,5,6,7,8,9,10,11,12,13,14,15);
}
__device__ __forceinline__ bf16x16 load_b_frag(const unsigned short* lds, int row, int half) {
    const unsigned short* p = lds + row * HSTR;
    bf16x8 lo = *(const bf16x8*)(p + 8 * half);
    bf16x8 hi = *(const bf16x8*)(p + 16 + 8 * half);
    return __builtin_shufflevector(lo, hi, 0,1,2,3,4,5,6,7,8,9,10,11,12,13,14,15);
}

// Fused hidden layer: pre = spk_in @ (W0+W1+W2)^T + bias, then LIF.
// spk_in: bf16 {0,1} [B,H].  Wp: 3 contiguous bf16 planes [3][H][H].
__global__ __launch_bounds__(256) void wmma_layer_lif_bf16_kernel(
        const unsigned short* __restrict__ spk_in,
        const unsigned short* __restrict__ Wp, int K,
        const float* __restrict__ bias,
        const float* __restrict__ beta,
        const float* __restrict__ thr,
        float* __restrict__ mem,
        unsigned short* __restrict__ spk_out,  // nullable
        float* __restrict__ rate_sum,          // nullable
        int H) {
    __shared__ unsigned short ldsA[BM * HSTR];        //  5,120 B
    __shared__ unsigned short ldsB[3 * BPLANE];       // 30,720 B
    const int m0  = blockIdx.x * BM;
    const int n0  = blockIdx.y * BN;
    const int tid = threadIdx.x;
    const size_t HH = (size_t)H * (size_t)H;

    v8f acc[2][2];
#pragma unroll
    for (int i = 0; i < 2; ++i)
#pragma unroll
        for (int j = 0; j < 2; ++j) acc[i][j] = (v8f){0.f,0.f,0.f,0.f,0.f,0.f,0.f,0.f};

    const int lane  = tid & 31;
    const int wave  = tid >> 5;
    const int wm    = wave & 1;
    const int wn    = wave >> 1;
    const int half  = lane >> 4;
    const int a0row = wm * 32 + (lane & 15);
    const int a1row = a0row + 16;
    const int b0row = wn * 32 + (lane & 15);
    const int b1row = b0row + 16;

    for (int k0 = 0; k0 < K; k0 += BKH) {
        __syncthreads();
        {   // A tile: 64 x 32 bf16 = 256 x 16B chunks, one per thread
            const int row = tid >> 2, c8 = (tid & 3) << 3;
            const unsigned short* src = spk_in + (size_t)(m0 + row) * (size_t)H + k0 + c8;
            copy16(src, ldsA + row * HSTR + c8);
            if (k0 + BKH < K) __builtin_prefetch(src + BKH, 0, 0);
        }
#pragma unroll
        for (int q = 0; q < 3; ++q) {   // W planes: 128 x 32 bf16 each
            const unsigned short* Wq = Wp + (size_t)q * HH;
            unsigned short* ldsq = ldsB + q * BPLANE;
#pragma unroll
            for (int i = 0; i < 2; ++i) {
                const int id = tid + (i << 8);
                const int row = id >> 2, c8 = (id & 3) << 3;
                copy16(Wq + (size_t)(n0 + row) * (size_t)H + k0 + c8,
                       ldsq + row * HSTR + c8);
            }
        }
        async_wait_zero();
        __syncthreads();

        const bf16x16 a0 = load_a_frag(ldsA, a0row, half);
        const bf16x16 a1 = load_a_frag(ldsA, a1row, half);
#pragma unroll
        for (int q = 0; q < 3; ++q) {
            const unsigned short* ldsq = ldsB + q * BPLANE;
            const bf16x16 b0 = load_b_frag(ldsq, b0row, half);
            const bf16x16 b1 = load_b_frag(ldsq, b1row, half);
            acc[0][0] = __builtin_amdgcn_wmma_f32_16x16x32_bf16(false, a0, false, b0,
                                                                (short)0, acc[0][0], false, false);
            acc[0][1] = __builtin_amdgcn_wmma_f32_16x16x32_bf16(false, a0, false, b1,
                                                                (short)0, acc[0][1], false, false);
            acc[1][0] = __builtin_amdgcn_wmma_f32_16x16x32_bf16(false, a1, false, b0,
                                                                (short)0, acc[1][0], false, false);
            acc[1][1] = __builtin_amdgcn_wmma_f32_16x16x32_bf16(false, a1, false, b1,
                                                                (short)0, acc[1][1], false, false);
        }
    }

    // LIF epilogue (reset-by-subtract), spikes stored as bf16 {0,1}.
#pragma unroll
    for (int ms = 0; ms < 2; ++ms) {
        const int mb = m0 + wm * 32 + ms * 16 + (half << 3);
#pragma unroll
        for (int ns = 0; ns < 2; ++ns) {
            const int nc = n0 + wn * 32 + ns * 16 + (lane & 15);
            const float bi = bias[nc], be = beta[nc], th = thr[nc];
#pragma unroll
            for (int r = 0; r < 8; ++r) {
                const size_t idx = (size_t)(mb + r) * (size_t)H + nc;
                float m_  = mem[idx];
                float rst = (m_ - th) > 0.0f ? 1.0f : 0.0f;
                float mn  = be * m_ + (acc[ms][ns][r] + bi) - rst * th;
                bool  s   = (mn - th) > 0.0f;
                mem[idx] = mn;
                if (spk_out)  spk_out[idx] = s ? (unsigned short)0x3F80 : (unsigned short)0;
                if (rate_sum) rate_sum[idx] += s ? 1.0f : 0.0f;
            }
        }
    }
}

// Elementwise input-layer LIF; emits bf16 spikes.
__global__ __launch_bounds__(256) void lif_input_kernel(
        const float* __restrict__ h_in,
        const float* __restrict__ beta,
        const float* __restrict__ thr,
        float* __restrict__ mem,
        unsigned short* __restrict__ spk_out, size_t n) {
    const size_t stride = (size_t)gridDim.x * blockDim.x;
    for (size_t i = (size_t)blockIdx.x * blockDim.x + threadIdx.x; i < n; i += stride) {
        const int h = (int)(i & 511);   // H = 512
        float m_  = mem[i];
        float th  = thr[h];
        float rst = (m_ - th) > 0.0f ? 1.0f : 0.0f;
        float mn  = beta[h] * m_ + h_in[i] - rst * th;
        mem[i]     = mn;
        spk_out[i] = (mn - th) > 0.0f ? (unsigned short)0x3F80 : (unsigned short)0;
    }
}

// out[b] = dot(rate_sum[b,:], W_out) * (1/T) + b_out.  One wave32 per row.
__global__ __launch_bounds__(256) void head_kernel(
        const float* __restrict__ rate_sum,
        const float* __restrict__ Wout,
        const float* __restrict__ bout,
        float* __restrict__ out, int H, float invT) {
    const int lane = threadIdx.x & 31;
    const int wave = threadIdx.x >> 5;
    const int b    = blockIdx.x * 8 + wave;
    const float* row = rate_sum + (size_t)b * (size_t)H;
    float s = 0.0f;
    for (int j = lane; j < H; j += 32) s += row[j] * Wout[j];
#pragma unroll
    for (int off = 16; off > 0; off >>= 1) s += __shfl_xor(s, off, 32);
    if (lane == 0) out[b] = s * invT + bout[0];
}

__global__ __launch_bounds__(256) void zero_kernel(float* __restrict__ p, size_t n) {
    const size_t stride = (size_t)gridDim.x * blockDim.x;
    for (size_t i = (size_t)blockIdx.x * blockDim.x + threadIdx.x; i < n; i += stride)
        p[i] = 0.0f;
}

extern "C" void kernel_launch(void* const* d_in, const int* in_sizes, int n_in,
                              void* d_out, int out_size, void* d_ws, size_t ws_size,
                              hipStream_t stream) {
    (void)in_sizes; (void)n_in; (void)out_size; (void)ws_size;
    const float* state   = (const float*)d_in[0];   // [B, 128]
    const float* action  = (const float*)d_in[1];   // [B, 16]
    const float* W_in    = (const float*)d_in[2];   // [512, 144]
    const float* b_in    = (const float*)d_in[3];   // [512]
    const float* beta_in = (const float*)d_in[4];   // [512]
    const float* thr_in  = (const float*)d_in[5];   // [512]
    const float* W_h     = (const float*)d_in[6];   // [2, 512, 512]
    const float* b_h     = (const float*)d_in[7];   // [2, 512]
    const float* beta_h  = (const float*)d_in[8];   // [2, 512]
    const float* thr_h   = (const float*)d_in[9];   // [2, 512]
    const float* W_out   = (const float*)d_in[10];  // [1, 512]
    const float* b_out   = (const float*)d_in[11];  // [1]
    float* out = (float*)d_out;

    const int B = 16384, S = 128, A = 16, H = 512, T = 8;
    const size_t BH = (size_t)B * (size_t)H;
    const size_t HH = (size_t)H * (size_t)H;

    float* ws     = (float*)d_ws;
    float* h_in   = ws;            // [B,H] f32
    float* mem_in = ws + 1 * BH;   // [B,H] f32
    float* mem1   = ws + 2 * BH;   // [B,H] f32
    float* mem2   = ws + 3 * BH;   // [B,H] f32
    float* rate   = ws + 4 * BH;   // [B,H] f32
    unsigned short* spkA   = (unsigned short*)(ws + 5 * BH);              // [B,H] bf16
    unsigned short* spkB   = spkA + BH;                                   // [B,H] bf16
    unsigned short* wsplit = spkB + BH;                                   // [2][3][H][H] bf16

    const dim3 blk(256);
    const dim3 ggrid(B / BM, H / BN);   // 256 x 4

    // Deterministic per-call init of all stateful buffers.
    zero_kernel<<<4096, blk, 0, stream>>>(mem_in, 3 * BH);   // mem_in, mem1, mem2
    zero_kernel<<<4096, blk, 0, stream>>>(rate, BH);

    // Exact 3-term bf16 split of the hidden-layer weights.
    split_w_kernel<<<2048, blk, 0, stream>>>(W_h, wsplit, HH, 2 * HH);

    // h_in = concat(state, action) @ W_in^T + b_in (exact fp32 WMMA path).
    wmma_gemm_bias_kernel<<<ggrid, blk, 0, stream>>>(state,  S, W_in,     S + A, S,
                                                     b_in,    h_in, H, 0);
    wmma_gemm_bias_kernel<<<ggrid, blk, 0, stream>>>(action, A, W_in + S, S + A, A,
                                                     nullptr, h_in, H, 1);

    for (int t = 0; t < T; ++t) {
        lif_input_kernel<<<4096, blk, 0, stream>>>(h_in, beta_in, thr_in,
                                                   mem_in, spkA, BH);
        wmma_layer_lif_bf16_kernel<<<ggrid, blk, 0, stream>>>(
            spkA, wsplit, H, b_h, beta_h, thr_h, mem1, spkB, nullptr, H);
        wmma_layer_lif_bf16_kernel<<<ggrid, blk, 0, stream>>>(
            spkB, wsplit + 3 * HH, H, b_h + H, beta_h + H, thr_h + H,
            mem2, nullptr, rate, H);
    }

    head_kernel<<<B / 8, blk, 0, stream>>>(rate, W_out, b_out, out, H, 1.0f / T);
}
